// MultiTaskUNetWithRegionDetection_71262097375867
// MI455X (gfx1250) — matrix operand order repaired and checked by reading, not compiled, not run on hardware
//
#include <hip/hip_runtime.h>
#include <cstdint>

// ---------------------------------------------------------------------------
// CDNA5 (gfx1250) implementation of MultiTaskUNetWithRegionDetection.
// All 3x3 convs run as implicit-GEMM on v_wmma_f32_16x16x32_f16 (wave32).
// - Activations stored f16 in channel-blocked NHWC32: [(C/32)][pixel][32ch].
// - Operand roles: A = weights (M = out-channels), B = activations
//   (N = pixels). A B-fragment is 16 consecutive channels at one pixel ->
//   ONE contiguous v16h load (2x global_load_b128) per 32-K chunk.
// - Weights pre-packed to f16 in exact A-fragment lane order (v16h loads).
// - Wave tile = 64 out-channels x 16 pixels: one B fragment feeds 4 WMMAs.
// - D: lane holds 8 consecutive channels at one pixel -> one v8h store.
// - Nearest-upsample and skip-concat folded into the conv's B indexing.
// ---------------------------------------------------------------------------

typedef __attribute__((ext_vector_type(16))) _Float16 v16h;
typedef __attribute__((ext_vector_type(8)))  _Float16 v8h;
typedef __attribute__((ext_vector_type(8)))  float    v8f;

__device__ __forceinline__ float sigmoidf_(float x) {
    return 1.0f / (1.0f + __expf(-x));
}

#define WMMA_F16(A, B, C) \
    __builtin_amdgcn_wmma_f32_16x16x32_f16(false, (A), false, (B), (short)0, (C), false, false)

// ---------------------------------------------------------------------------
// Weight pre-pack: OIHW f32 -> f16 in A-fragment order.
// Fragment (nt, r, cbi) is a 512-element block; element (lane, h) holds
//   w[n = nt*16 + lane%16]
//    [ci = cbi*32 + h + (lane>=16 ? 16 : 0)][tap r].
// ---------------------------------------------------------------------------
__global__ void pack_weights_k(const float* __restrict__ w,
                               _Float16* __restrict__ wp,
                               int Cout, int Cin)
{
    const long tot = (long)Cout * Cin * 9;
    long idx = (long)blockIdx.x * blockDim.x + threadIdx.x;
    if (idx >= tot) return;
    const int  Cin32 = Cin >> 5;
    int  h    = (int)(idx & 15);
    int  lane = (int)((idx >> 4) & 31);
    long rest = idx >> 9;
    int  cbi  = (int)(rest % Cin32);
    long rst2 = rest / Cin32;
    int  r    = (int)(rst2 % 9);
    int  nt   = (int)(rst2 / 9);
    int  n    = nt * 16 + (lane & 15);
    int  ci   = cbi * 32 + h + ((lane >> 4) ? 16 : 0);
    wp[idx] = (_Float16)w[((long)n * Cin + ci) * 9 + r];
}

// ---------------------------------------------------------------------------
// Fast conv path: 3x3, pad=1, stride=1, Cin % 32 == 0, Cout % 64 == 0.
// NHWC32 f16 in / NHWC32 f16 out. Optional folded x2 nearest upsample (up2)
// and folded concat (in0 = blocks [0,C0b), in1 = blocks [C0b,Cin/32)).
// act: 0=none, 1=relu, 2=sigmoid.
// Wave tile: 64 channels x 16 pixels. grid = (npix/64, Cout/64), block 128.
// ---------------------------------------------------------------------------
__global__ __launch_bounds__(128) void conv3x3_fast_k(
    const _Float16* __restrict__ in0, int C0b,
    const _Float16* __restrict__ in1,
    const _Float16* __restrict__ wp, const float* __restrict__ bias,
    _Float16* __restrict__ out,
    int Cin, int Hin, int Win, int up2, int act)
{
    const int lane  = threadIdx.x & 31;
    const int wv    = threadIdx.x >> 5;
    const int lhalf = lane & 15;
    const int hi    = lane >> 4;
    const int Hout  = up2 ? (Hin << 1) : Hin;
    const int Wout  = up2 ? (Win << 1) : Win;
    const int npix  = Hout * Wout;
    const int m0    = (blockIdx.x * 4 + wv) * 16;   // pixel tile base
    const int nt0   = blockIdx.y * 4;               // 4 n-tiles = 64 channels
    const int Cin32 = Cin >> 5;
    const long spin32   = (long)Hin * Win * 32;     // NHWC32 block stride
    const long ntstride = 9L * Cin32 * 512;         // packed n-tile stride

    const int oy = m0 / Wout;                       // uniform (row segment)
    const int ox = (m0 - oy * Wout) + lhalf;        // lane pixel x
    const int boff16 = hi ? 16 : 0;                 // B half-wave K offset
    const int lane16 = lane * 16;                   // A-fragment lane offset

    v8f acc[4] = {};

#pragma unroll
    for (int r = 0; r < 9; ++r) {
        const int  kh = r / 3, kw = r - kh * 3;     // constants after unroll
        const int  iy  = oy + kh - 1;               // uniform
        const bool yok = (iy >= 0) & (iy < Hout);   // uniform -> scalar branch
        const int  ix  = ox + kw - 1;               // per-lane
        const bool vmask = yok & (ix >= 0) & (ix < Wout);
        const int  sy = up2 ? (iy >> 1) : iy;
        const int  sx = up2 ? (ix >> 1) : ix;
        const int  pixoff = (sy * Win + sx) * 32 + boff16;  // per-lane i32

        const _Float16* wt0 = wp + ((long)nt0 * 9 + r) * Cin32 * 512;
        const _Float16* wt1 = wt0 + ntstride;
        const _Float16* wt2 = wt0 + 2 * ntstride;
        const _Float16* wt3 = wt0 + 3 * ntstride;

        // B fragment zero-init ONCE per tap: invalid lanes never load and
        // stay zero across the whole cbi loop (vmask is cbi-invariant).
        v16h bfrag;
#pragma unroll
        for (int h = 0; h < 16; ++h) bfrag[h] = (_Float16)0.0f;

        for (int cbi = 0; cbi < Cin32; ++cbi) {
            // ---- B: 16 consecutive channels at lane's pixel: 1 v16h load --
            if (vmask) {
                const _Float16* src = (cbi < C0b)
                    ? (in0 + (long)cbi * spin32)
                    : (in1 + (long)(cbi - C0b) * spin32);   // uniform select
                bfrag = *reinterpret_cast<const v16h*>(src + pixoff);
            }
            // ---- A: 4 packed weight fragments: uniform base + lane*32B ----
            const long co = (long)cbi * 512;
            __builtin_prefetch((const void*)(wt0 + co + 512), 0, 1);
            v16h a0 = *reinterpret_cast<const v16h*>(wt0 + co + lane16);
            v16h a1 = *reinterpret_cast<const v16h*>(wt1 + co + lane16);
            v16h a2 = *reinterpret_cast<const v16h*>(wt2 + co + lane16);
            v16h a3 = *reinterpret_cast<const v16h*>(wt3 + co + lane16);
            acc[0] = WMMA_F16(a0, bfrag, acc[0]);
            acc[1] = WMMA_F16(a1, bfrag, acc[1]);
            acc[2] = WMMA_F16(a2, bfrag, acc[2]);
            acc[3] = WMMA_F16(a3, bfrag, acc[3]);
        }
    }

    // ---- D: lane = pixel, rows = 8 consecutive channels -> one v8h store --
    const int hi8 = hi ? 8 : 0;
    const int pix = m0 + lhalf;
#pragma unroll
    for (int t = 0; t < 4; ++t) {
        const int cg   = (nt0 + t) * 16;            // tile channel base
        const int cblk = cg >> 5;                   // NHWC32 block
        const int cwi  = (cg & 31) + hi8;           // offset within block
        v8f bv = *reinterpret_cast<const v8f*>(bias + cg + hi8);
        _Float16* ob = out + ((long)cblk * npix + pix) * 32 + cwi;
        v8h sv;
#pragma unroll
        for (int rr = 0; rr < 8; ++rr) {
            float v = acc[t][rr] + bv[rr];
            if (act == 1)      v = fmaxf(v, 0.0f);
            else if (act == 2) v = sigmoidf_(v);
            sv[rr] = (_Float16)v;
        }
        *reinterpret_cast<v8h*>(ob) = sv;
    }
}

// ---------------------------------------------------------------------------
// Generic conv path (Cin = 1 or 6; handles stride 2). Planar f32 in,
// NHWC32 f16 out (scatter stores; only 2 cheap layers use this).
// GEMM roles: M = pixels, N = channels (original orientation).
// ---------------------------------------------------------------------------
__global__ __launch_bounds__(128) void conv3x3_gen_k(
    const float* __restrict__ in0,
    const float* __restrict__ w, const float* __restrict__ bias,
    _Float16* __restrict__ out,
    int Cin, int Hin, int Win, int Hout, int Wout, int stride, int act)
{
    const int lane  = threadIdx.x & 31;
    const int wv    = threadIdx.x >> 5;
    const int lhalf = lane & 15;
    const int hi    = lane >> 4;
    const int npix  = Hout * Wout;
    const int m0    = (blockIdx.x * 4 + wv) * 16;
    const int n0    = blockIdx.y * 32;
    const int K     = Cin * 9;
    const long spin = (long)Hin * Win;

    const int oy = m0 / Wout;              // uniform (tile = row segment)
    const int ox = (m0 - oy * Wout) + lhalf;
    const int aoff = hi ? 8 : 0;
    const int boff = hi ? 16 : 0;

    v8f acc0 = {};
    v8f acc1 = {};

    for (int kk = 0; kk < K; kk += 32) {
        v16h afrag, b0, b1;
#pragma unroll
        for (int h = 0; h < 16; ++h) {
            int k = kk + h + boff;
            float va = 0.0f;
            if (k < K) {
                int ci = k / 9;
                int r  = k - ci * 9;
                int iy = oy * stride + (r / 3) - 1;
                int ix = ox * stride + (r - (r / 3) * 3) - 1;
                if (iy >= 0 && iy < Hin && ix >= 0 && ix < Win)
                    va = in0[(long)ci * spin + (long)iy * Win + ix];
            }
            afrag[h] = (_Float16)va;
        }
#pragma unroll
        for (int h = 0; h < 16; ++h) {
            int k = kk + h + boff;
            float v0 = 0.0f, v1 = 0.0f;
            if (k < K) {
                v0 = w[(long)(n0 + lhalf) * K + k];
                v1 = w[(long)(n0 + 16 + lhalf) * K + k];
            }
            b0[h] = (_Float16)v0;
            b1[h] = (_Float16)v1;
        }
        acc0 = WMMA_F16(afrag, b0, acc0);
        acc1 = WMMA_F16(afrag, b1, acc1);
    }

    // scatter-store into NHWC32 (lane = channel, rows = pixels)
    const int   na  = n0 + lhalf;
    const int   nb  = n0 + 16 + lhalf;
    const float bv0 = bias[na];
    const float bv1 = bias[nb];
    const int   hi8 = hi ? 8 : 0;
    const long  baseA = ((long)(na >> 5) * npix) * 32 + (na & 31);
    const long  baseB = ((long)(nb >> 5) * npix) * 32 + (nb & 31);
#pragma unroll
    for (int rr = 0; rr < 8; ++rr) {
        const int pix = m0 + rr + hi8;
        float v0 = acc0[rr] + bv0;
        float v1 = acc1[rr] + bv1;
        if (act == 1)      { v0 = fmaxf(v0, 0.0f);  v1 = fmaxf(v1, 0.0f); }
        else if (act == 2) { v0 = sigmoidf_(v0);    v1 = sigmoidf_(v1); }
        out[baseA + (long)pix * 32] = (_Float16)v0;
        out[baseB + (long)pix * 32] = (_Float16)v1;
    }
}

// ---------------------------------------------------------------------------
// 2x2 max pool, stride 2, NHWC32 f16 -> NHWC32 f16.
// ---------------------------------------------------------------------------
__global__ void maxpool2_k(const _Float16* __restrict__ in,
                           _Float16* __restrict__ out,
                           int C, int Hin, int Win)
{
    const int Hout = Hin >> 1, Wout = Win >> 1;
    const long tot = (long)C * Hout * Wout;
    long idx = (long)blockIdx.x * blockDim.x + threadIdx.x;
    if (idx >= tot) return;
    int c32 = (int)(idx & 31);
    long r2 = idx >> 5;
    int x   = (int)(r2 % Wout);
    long r3 = r2 / Wout;
    int y   = (int)(r3 % Hout);
    int cb  = (int)(r3 / Hout);
    const _Float16* p = in + ((long)cb * Hin * Win + (long)(2 * y) * Win + 2 * x) * 32 + c32;
    const long wrow = (long)Win * 32;
    float v = fmaxf(fmaxf((float)p[0], (float)p[32]),
                    fmaxf((float)p[wrow], (float)p[wrow + 32]));
    out[idx] = (_Float16)v;
}

// ---------------------------------------------------------------------------
// 1x1 conv head: 64 -> 6 channels (NHWC32 f16 in, planar f32 out).
// ---------------------------------------------------------------------------
__global__ void conv1x1_head_k(const _Float16* __restrict__ in,
                               const float* __restrict__ w,
                               const float* __restrict__ b,
                               float* __restrict__ out,
                               int Cin, int npix, int act)
{
    int p = blockIdx.x * blockDim.x + threadIdx.x;
    if (p >= npix) return;
    float s[6];
#pragma unroll
    for (int c = 0; c < 6; ++c) s[c] = b[c];
    const int nblk = Cin >> 5;
    for (int cb = 0; cb < nblk; ++cb) {
        const _Float16* ip = in + ((long)cb * npix + p) * 32;
        for (int c32 = 0; c32 < 32; ++c32) {
            float v = (float)ip[c32];
            const int ci = cb * 32 + c32;
#pragma unroll
            for (int c = 0; c < 6; ++c) s[c] += v * w[c * Cin + ci];
        }
    }
#pragma unroll
    for (int c = 0; c < 6; ++c) {
        float v = s[c];
        if (act == 2) v = sigmoidf_(v);
        out[(long)c * npix + p] = v;
    }
}

// ---------------------------------------------------------------------------
// Per-channel spatial mean: one block per channel (planar f32 in).
// ---------------------------------------------------------------------------
__global__ void chan_mean_k(const float* __restrict__ in, float* __restrict__ out,
                            int npix)
{
    __shared__ float red[256];
    const int c = blockIdx.x;
    float s = 0.0f;
    for (int i = threadIdx.x; i < npix; i += 256)
        s += in[(long)c * npix + i];
    red[threadIdx.x] = s;
    __syncthreads();
    for (int off = 128; off > 0; off >>= 1) {
        if ((int)threadIdx.x < off) red[threadIdx.x] += red[threadIdx.x + off];
        __syncthreads();
    }
    if (threadIdx.x == 0) out[c] = red[0] / (float)npix;
}

// ---------------------------------------------------------------------------
// bbox head: bb[j] = sigmoid(pooled . W[j] + b[j]) * 512,  j in [0,20)
// ---------------------------------------------------------------------------
__global__ void bbox_head_k(const float* __restrict__ pooled,
                            const float* __restrict__ w,
                            const float* __restrict__ b,
                            float* __restrict__ bb)
{
    int j = threadIdx.x;
    if (j < 20) {
        float s = b[j];
#pragma unroll
        for (int c = 0; c < 6; ++c) s += pooled[c] * w[j * 6 + c];
        bb[j] = sigmoidf_(s) * 512.0f;
    }
}

// ---------------------------------------------------------------------------
// Rasterize 5 bboxes into a (6,H,W) one-hot map (channel 0 = zeros);
// writes the two duplicated f32 output copies directly.
// ---------------------------------------------------------------------------
__global__ void raster_onehot_k(const float* __restrict__ bb,
                                float* __restrict__ out1,
                                float* __restrict__ out2,
                                int H, int W)
{
    const long tot = 6L * H * W;
    long idx = (long)blockIdx.x * blockDim.x + threadIdx.x;
    if (idx >= tot) return;
    const int hw = H * W;
    int c   = (int)(idx / hw);
    int rem = (int)(idx - (long)c * hw);
    int y   = rem / W;
    int x   = rem - y * W;
    float v = 0.0f;
    if (c >= 1) {
        int o = c - 1;
        int xmin = (int)bb[o * 4 + 0];
        int ymin = (int)bb[o * 4 + 1];
        int xmax = (int)bb[o * 4 + 2];
        int ymax = (int)bb[o * 4 + 3];
        v = (y >= ymin && y < ymax && x >= xmin && x < xmax) ? 1.0f : 0.0f;
    }
    out1[idx] = v;
    out2[idx] = v;
}

// ---------------------------------------------------------------------------
// Bilinear resize (align_corners=True) + sigmoid, NHWC32 f16 -> NHWC32 f16.
// ---------------------------------------------------------------------------
__global__ void resize_bilinear_sig_k(const _Float16* __restrict__ in,
                                      _Float16* __restrict__ out,
                                      int C, int Hin, int Win, int Hout, int Wout)
{
    const long tot = (long)C * Hout * Wout;
    long idx = (long)blockIdx.x * blockDim.x + threadIdx.x;
    if (idx >= tot) return;
    int c32 = (int)(idx & 31);
    long r2 = idx >> 5;
    int x   = (int)(r2 % Wout);
    long r3 = r2 / Wout;
    int y   = (int)(r3 % Hout);
    int cb  = (int)(r3 / Hout);
    float fy = (float)y * (float)(Hin - 1) / (float)(Hout - 1);
    float fx = (float)x * (float)(Win - 1) / (float)(Wout - 1);
    int y0 = (int)floorf(fy); int y1 = (y0 + 1 < Hin) ? y0 + 1 : Hin - 1;
    int x0 = (int)floorf(fx); int x1 = (x0 + 1 < Win) ? x0 + 1 : Win - 1;
    float ty = fy - (float)y0, tx = fx - (float)x0;
    const _Float16* p = in + (long)cb * Hin * Win * 32 + c32;
    float v00 = (float)p[((long)y0 * Win + x0) * 32];
    float v01 = (float)p[((long)y0 * Win + x1) * 32];
    float v10 = (float)p[((long)y1 * Win + x0) * 32];
    float v11 = (float)p[((long)y1 * Win + x1) * 32];
    float v = (v00 * (1.0f - tx) + v01 * tx) * (1.0f - ty)
            + (v10 * (1.0f - tx) + v11 * tx) * ty;
    out[idx] = (_Float16)sigmoidf_(v);
}

// ---------------------------------------------------------------------------
// Host-side launchers.
// ---------------------------------------------------------------------------
static void conv3x3_fast(hipStream_t s,
                         const _Float16* in0, int C0, const _Float16* in1,
                         const _Float16* wp, const float* b, _Float16* out,
                         int Cin, int Cout, int Hin, int Win, int up2, int act)
{
    int Hout = up2 ? Hin * 2 : Hin;
    int Wout = up2 ? Win * 2 : Win;
    dim3 grid((unsigned)((Hout * Wout) / 64), (unsigned)(Cout / 64));
    conv3x3_fast_k<<<grid, 128, 0, s>>>(in0, C0 >> 5, in1, wp, b, out,
                                        Cin, Hin, Win, up2, act);
}

static void conv3x3_gen(hipStream_t s,
                        const float* in0, const float* w, const float* b,
                        _Float16* out, int Cin, int Cout, int Hin, int Win,
                        int stride, int act)
{
    int Hout = Hin / stride;
    int Wout = Win / stride;
    dim3 grid((unsigned)((Hout * Wout) / 64), (unsigned)(Cout / 32));
    conv3x3_gen_k<<<grid, 128, 0, s>>>(in0, w, b, out,
                                       Cin, Hin, Win, Hout, Wout, stride, act);
}

extern "C" void kernel_launch(void* const* d_in, const int* in_sizes, int n_in,
                              void* d_out, int out_size, void* d_ws, size_t ws_size,
                              hipStream_t stream)
{
    (void)in_sizes; (void)n_in; (void)out_size; (void)ws_size;

    const float* x = (const float*)d_in[0];
    auto P = [&](int i) { return (const float*)d_in[i]; };

    const int HW  = 512 * 512;
    const int HW2 = 256 * 256;
    const int HW3 = 128 * 128;
    const int HW4 = 64 * 64;

    // ---- workspace bump allocator ----
    float* ws = (float*)d_ws;
    size_t off = 0;
    auto allocf = [&](size_t n) { float* p = ws + off; off += n; return p; };
    auto alloch = [&](size_t n) { return (_Float16*)allocf((n + 1) / 2); };

    _Float16* f1     = alloch((size_t)64  * HW);   // encoder skip 1
    _Float16* f2     = alloch((size_t)128 * HW2);  // encoder skip 2
    _Float16* f3     = alloch((size_t)256 * HW3);  // encoder skip 3
    _Float16* f4     = alloch((size_t)512 * HW4);  // encoder final
    _Float16* t512a  = alloch((size_t)64  * HW);   // 512^2 temporaries
    _Float16* t512b  = alloch((size_t)64  * HW);
    _Float16* ds_out = t512a;   // alias: 512*HW2 f16 == t512a+t512b, disjoint live range
    _Float16* t256a  = alloch((size_t)128 * HW2);
    _Float16* t256b  = alloch((size_t)128 * HW2);
    _Float16* t128a  = alloch((size_t)256 * HW3);
    _Float16* t128b  = alloch((size_t)256 * HW3);
    _Float16* t64a   = alloch((size_t)512 * HW4);
    _Float16* p256   = alloch((size_t)64  * HW2);  // pooled encoder tensors
    _Float16* p128   = alloch((size_t)128 * HW3);
    _Float16* p64    = alloch((size_t)256 * HW4);
    _Float16* act64  = alloch((size_t)512 * HW4);  // sigmoid(resized ds)
    float*    first  = allocf((size_t)6 * HW);     // sigmoid(d1 output), f32
    float*    pooled = allocf(8);
    float*    bb     = allocf(32);

    // ---- pack f32 OIHW weights into f16 A-fragment order ----
    auto pack = [&](const float* w, int Cout, int Cin) {
        size_t n = (size_t)Cout * Cin * 9;
        _Float16* p = alloch(n);
        pack_weights_k<<<(unsigned)((n + 255) / 256), 256, 0, stream>>>(w, p, Cout, Cin);
        return p;
    };

    _Float16* pk_e1b = pack(P(3),  64,  64);
    _Float16* pk_e2a = pack(P(5),  128, 64);
    _Float16* pk_e2b = pack(P(7),  128, 128);
    _Float16* pk_e3a = pack(P(9),  256, 128);
    _Float16* pk_e3b = pack(P(11), 256, 256);
    _Float16* pk_e4a = pack(P(13), 512, 256);
    _Float16* pk_e4b = pack(P(15), 512, 512);
    _Float16* dpk[2][9];
    const int bases[2] = {17, 37};
    for (int d = 0; d < 2; ++d) {
        int b = bases[d];
        dpk[d][0] = pack(P(b + 0),  256, 512);  // up3
        dpk[d][1] = pack(P(b + 2),  256, 512);  // c3a
        dpk[d][2] = pack(P(b + 4),  256, 256);  // c3b
        dpk[d][3] = pack(P(b + 6),  128, 256);  // up2
        dpk[d][4] = pack(P(b + 8),  128, 256);  // c2a
        dpk[d][5] = pack(P(b + 10), 128, 128);  // c2b
        dpk[d][6] = pack(P(b + 12), 64,  128);  // up1
        dpk[d][7] = pack(P(b + 14), 64,  128);  // c1a
        dpk[d][8] = pack(P(b + 16), 64,  64);   // c1b
    }

    float* out         = (float*)d_out;
    const long SEG     = 6L * HW;
    float* out_onehot1 = out + SEG;
    float* out_onehot2 = out + 2 * SEG;

    // ================= encoder =================
    conv3x3_gen (stream, x, P(1), P(2), t512a, 1, 64, 512, 512, 1, 1);
    conv3x3_fast(stream, t512a, 64, nullptr, pk_e1b, P(4), f1, 64, 64, 512, 512, 0, 1);
    maxpool2_k<<<(64 * HW2 + 255) / 256, 256, 0, stream>>>(f1, p256, 64, 512, 512);

    conv3x3_fast(stream, p256,  64,  nullptr, pk_e2a, P(6),  t256a, 64,  128, 256, 256, 0, 1);
    conv3x3_fast(stream, t256a, 128, nullptr, pk_e2b, P(8),  f2,    128, 128, 256, 256, 0, 1);
    maxpool2_k<<<(128 * HW3 + 255) / 256, 256, 0, stream>>>(f2, p128, 128, 256, 256);

    conv3x3_fast(stream, p128,  128, nullptr, pk_e3a, P(10), t128a, 128, 256, 128, 128, 0, 1);
    conv3x3_fast(stream, t128a, 256, nullptr, pk_e3b, P(12), f3,    256, 256, 128, 128, 0, 1);
    maxpool2_k<<<(256 * HW4 + 255) / 256, 256, 0, stream>>>(f3, p64, 256, 128, 128);

    conv3x3_fast(stream, p64,  256, nullptr, pk_e4a, P(14), t64a, 256, 512, 64, 64, 0, 1);
    conv3x3_fast(stream, t64a, 512, nullptr, pk_e4b, P(16), f4,   512, 512, 64, 64, 0, 1);

    // ================= decoder (shared for d1/d2) =================
    auto run_decoder = [&](int d, const _Float16* hin, float* head_out, int head_act) {
        const int b = bases[d];
        // level 3: up(512->256 @128^2), concat with f3, 2x conv
        conv3x3_fast(stream, hin,   512, nullptr, dpk[d][0], P(b + 1),  t128a, 512, 256, 64,  64,  1, 1);
        conv3x3_fast(stream, f3,    256, t128a,   dpk[d][1], P(b + 3),  t128b, 512, 256, 128, 128, 0, 1);
        conv3x3_fast(stream, t128b, 256, nullptr, dpk[d][2], P(b + 5),  t128a, 256, 256, 128, 128, 0, 1);
        // level 2
        conv3x3_fast(stream, t128a, 256, nullptr, dpk[d][3], P(b + 7),  t256a, 256, 128, 128, 128, 1, 1);
        conv3x3_fast(stream, f2,    128, t256a,   dpk[d][4], P(b + 9),  t256b, 256, 128, 256, 256, 0, 1);
        conv3x3_fast(stream, t256b, 128, nullptr, dpk[d][5], P(b + 11), t256a, 128, 128, 256, 256, 0, 1);
        // level 1
        conv3x3_fast(stream, t256a, 128, nullptr, dpk[d][6], P(b + 13), t512a, 128, 64, 256, 256, 1, 1);
        conv3x3_fast(stream, f1,    64,  t512a,   dpk[d][7], P(b + 15), t512b, 128, 64, 512, 512, 0, 1);
        conv3x3_fast(stream, t512b, 64,  nullptr, dpk[d][8], P(b + 17), t512a, 64,  64, 512, 512, 0, 1);
        // 1x1 head
        conv1x1_head_k<<<(HW + 255) / 256, 256, 0, stream>>>(
            t512a, P(b + 18), P(b + 19), head_out, 64, HW, head_act);
    };

    // ---- decoder 1 + region head ----
    run_decoder(0, f4, first, /*sigmoid*/ 2);
    chan_mean_k<<<6, 256, 0, stream>>>(first, pooled, HW);
    bbox_head_k<<<1, 32, 0, stream>>>(pooled, P(57), P(58), bb);
    raster_onehot_k<<<(int)((SEG + 255) / 256), 256, 0, stream>>>(
        bb, out_onehot1, out_onehot2, 512, 512);

    // ---- downsample conv (6->512, stride 2) + bilinear resize + sigmoid ----
    conv3x3_gen(stream, out_onehot1, P(59), P(60), ds_out, 6, 512, 512, 512, 2, 1);
    resize_bilinear_sig_k<<<(512 * HW4 + 255) / 256, 256, 0, stream>>>(
        ds_out, act64, 512, 256, 256, 64, 64);

    // ---- decoder 2 -> segmentation output (no activation) ----
    run_decoder(1, act64, out, /*none*/ 0);
}
